// DiTBlockFinal_81252191306508
// MI455X (gfx1250) — matrix operand order repaired and checked by reading, not compiled, not run on hardware
//
#include <hip/hip_runtime.h>
#include <hip/hip_bf16.h>

// ---------------------------------------------------------------------------
// DiT block for MI455X (gfx1250, wave32, WMMA + Tensor Data Mover).
// f32 in/out; all matrix operands held in f16; WMMA f16 -> f32 accum.
// Tile staging global->LDS goes through tensor_load_to_lds (TENSORcnt).
// ---------------------------------------------------------------------------

typedef _Float16 half4_t  __attribute__((ext_vector_type(4)));
typedef _Float16 half8_t  __attribute__((ext_vector_type(8)));
typedef _Float16 v16h     __attribute__((ext_vector_type(16)));
typedef float    v8f      __attribute__((ext_vector_type(8)));
typedef unsigned int uint32x4 __attribute__((ext_vector_type(4)));
typedef int      int32x4  __attribute__((ext_vector_type(4)));
typedef int      int32x8  __attribute__((ext_vector_type(8)));

#define DIT_B      8
#define DIT_N      1024
#define DIT_H      1152
#define DIT_NH     16
#define DIT_HD     72
#define DIT_CTX    768
#define DIT_CTOK   77
#define DIT_MLP    4608
#define DIT_MODW   (6 * DIT_H)
#define DIT_MTOK   (DIT_B * DIT_N)      // 8192 token rows
#define DIT_CROWS  (DIT_B * DIT_CTOK)   // 616 context rows

// ---------------- WMMA fragment helpers (16x16x32 f16) ----------------------
__device__ __forceinline__ v16h load_frag_a(const _Float16* base, int ldh, int k0) {
  const int lane = threadIdx.x & 31;
  const int m    = lane & 15;
  const int hi   = lane >> 4;
  const _Float16* row = base + (size_t)m * ldh;
  union { v16h v; half8_t h[2]; } u;
  u.h[0] = *(const half8_t*)(row + k0 + hi * 8);
  u.h[1] = *(const half8_t*)(row + k0 + 16 + hi * 8);
  return u.v;
}

__device__ __forceinline__ v16h load_frag_b(const _Float16* base, int ldh, int k0) {
  const int lane = threadIdx.x & 31;
  const int n    = lane & 15;
  const int hi   = lane >> 4;
  const _Float16* row = base + (size_t)n * ldh;
  union { v16h v; half8_t h[2]; } u;
  u.h[0] = *(const half8_t*)(row + k0 + hi * 8);
  u.h[1] = *(const half8_t*)(row + k0 + 16 + hi * 8);
  return u.v;
}

__device__ __forceinline__ v8f wmma_f16(v16h a, v16h b, v8f c) {
  return __builtin_amdgcn_wmma_f32_16x16x32_f16(false, a, false, b, (short)0, c,
                                                false, false);
}

// ---------------- Tensor Data Mover: 2-D f16 tile -> LDS ---------------------
// D# per CDNA5 ISA ch.8: group0 = {count/lds_addr/global_addr/type},
// group1 = {flags, tensor_dim0/1, tile_dim0/1, tensor_dim0_stride}.
// tensor_dim zero-fills OOB reads; pad_enable skews LDS rows.
// Toolchain here exposes the 6-arg builtin (extra trailing int32x8 group).
__device__ __forceinline__ void tdm_load_2d_f16(
    unsigned lds_addr, const _Float16* gptr,
    unsigned tile0, unsigned tile1,          // tile dims (elements)
    unsigned td0, unsigned td1,              // tensor dims from tile origin
    unsigned long long stride0,              // row stride (elements)
    int pad_en, int pad_interval_code, int pad_amount_code) {
  unsigned long long ga = (unsigned long long)(uintptr_t)gptr;
  uint32x4 g0;
  g0[0] = 1u;                                   // count=1 (valid), user mode
  g0[1] = lds_addr;                             // lds_addr
  g0[2] = (unsigned)ga;                         // global_addr[31:0]
  g0[3] = (unsigned)((ga >> 32) & 0x01ffffffu)  // global_addr[56:32]
        | (2u << 30);                           // type=2 ("image")
  int32x8 g1;
  g1[0] = (1 << 16)                             // data_size = 2 bytes
        | (pad_en << 20)
        | (pad_interval_code << 22)
        | (pad_amount_code << 25);
  g1[1] = (int)((td0 & 0xffffu) << 16);                               // dim0 lo
  g1[2] = (int)(((td0 >> 16) & 0xffffu) | ((td1 & 0xffffu) << 16));   // dim0 hi|dim1 lo
  g1[3] = (int)(((td1 >> 16) & 0xffffu) | ((tile0 & 0xffffu) << 16)); // dim1 hi|tile0
  g1[4] = (int)(tile1 & 0xffffu);                                     // tile1, tile2=0
  g1[5] = (int)(stride0 & 0xffffffffull);                             // stride0 lo
  g1[6] = (int)((stride0 >> 32) & 0xffffull);                         // stride0 hi
  g1[7] = 0;
  int32x4 z4 = {0, 0, 0, 0};
  int32x8 z8 = {0, 0, 0, 0, 0, 0, 0, 0};
  __builtin_amdgcn_tensor_load_to_lds(g0, g1, z4, z4, z8, 0);
}

__device__ __forceinline__ unsigned lds_addr_of(const void* p) {
  return (unsigned)(uintptr_t)p;   // flat LDS aperture: low 32 bits = LDS offset
}

__device__ __forceinline__ float gelu_tanh(float v) {
  const float c = 0.7978845608028654f;  // sqrt(2/pi)
  float u = c * (v + 0.044715f * v * v * v);
  return 0.5f * v * (1.0f + tanhf(u));
}

// ---------------- f32 -> f16 conversion --------------------------------------
__global__ __launch_bounds__(256) void cvt_f16_kernel(const float* __restrict__ src,
                                                      _Float16* __restrict__ dst,
                                                      int n4) {   // n/4 groups
  int i = blockIdx.x * 256 + threadIdx.x;
  if (i < n4) {
    float4 f = *(const float4*)(src + (size_t)i * 4);
    half4_t h = { (_Float16)f.x, (_Float16)f.y, (_Float16)f.z, (_Float16)f.w };
    *(half4_t*)(dst + (size_t)i * 4) = h;
  }
}

// ---------------- adaLN modulation: mod = silu(t) @ ada_w^T + ada_b ---------
__global__ __launch_bounds__(256) void ada_kernel(const float* __restrict__ t,
                                                  const float* __restrict__ w,
                                                  const float* __restrict__ bias,
                                                  float* __restrict__ mod) {
  __shared__ float st[DIT_H];
  const int b = (blockIdx.x * 256) / DIT_MODW;      // 6912 % 256 == 0 -> uniform
  for (int i = threadIdx.x; i < DIT_H; i += 256) {
    float tv = t[b * DIT_H + i];
    st[i] = tv / (1.0f + __expf(-tv));
  }
  __syncthreads();
  const int j = blockIdx.x * 256 + threadIdx.x - b * DIT_MODW;
  const float* wr = w + (size_t)j * DIT_H;
  float acc = bias[j];
  for (int k = 0; k < DIT_H; ++k) acc += st[k] * wr[k];
  mod[(size_t)b * DIT_MODW + j] = acc;
}

// ---------------- LayerNorm + modulate (f32 in, f16 out) --------------------
__global__ __launch_bounds__(128) void ln_mod_kernel(const float* __restrict__ x,
                                                     const float* __restrict__ mod,
                                                     int sh_off, int sc_off,
                                                     _Float16* __restrict__ out) {
  const int row = blockIdx.x;          // 0..8191
  const int b   = row >> 10;
  const float* xr  = x + (size_t)row * DIT_H;
  const float* shp = mod + (size_t)b * DIT_MODW + sh_off;
  const float* scp = mod + (size_t)b * DIT_MODW + sc_off;
  float s = 0.f, s2 = 0.f;
  for (int i = threadIdx.x; i < DIT_H; i += 128) {
    float v = xr[i];
    s += v; s2 += v * v;
  }
  for (int m = 16; m; m >>= 1) {
    s  += __shfl_xor(s, m, 32);
    s2 += __shfl_xor(s2, m, 32);
  }
  __shared__ float rs[4], rs2[4];
  const int w = threadIdx.x >> 5;
  if ((threadIdx.x & 31) == 0) { rs[w] = s; rs2[w] = s2; }
  __syncthreads();
  s  = rs[0] + rs[1] + rs[2] + rs[3];
  s2 = rs2[0] + rs2[1] + rs2[2] + rs2[3];
  const float mu   = s * (1.0f / DIT_H);
  const float var  = s2 * (1.0f / DIT_H) - mu * mu;
  const float rstd = rsqrtf(var + 1e-6f);
  _Float16* orow = out + (size_t)row * DIT_H;
  for (int i = threadIdx.x; i < DIT_H; i += 128)
    orow[i] = (_Float16)((xr[i] - mu) * rstd * (1.0f + scp[i]) + shp[i]);
}

// ---------------- WMMA GEMM: C = A[M,K](f16) * W[N,K]^T(f16) + epilogue -----
#define EPI_NONE 0
#define EPI_GELU 1
#define EPI_RES  2   // C = resid + (gate? gate : 1) * (acc + bias)

#define GBM 128
#define GBN 128
#define GBK 64
#define GLD (GBK + 8)   // 72 halves = 144B (16B multiple), via TDM pad

__global__ __launch_bounds__(512) void gemm_kernel(
    const _Float16* __restrict__ A, const _Float16* __restrict__ W,
    const float* __restrict__ bias,
    float* __restrict__ C, _Float16* __restrict__ C16,
    int M, int N, int K, int epi,
    const float* __restrict__ resid, const float* __restrict__ gate,
    int gate_ld, int rows_per_batch) {
  __shared__ __attribute__((aligned(16))) _Float16 As[2][GBM][GLD];
  __shared__ __attribute__((aligned(16))) _Float16 Bs[2][GBN][GLD];

  const int tid  = threadIdx.x;
  const int lane = tid & 31;
  const int hi   = lane >> 4;
  const int ln   = lane & 15;
  const int w    = tid >> 5;           // 0..15
  const int wm   = w >> 2;
  const int wn   = w & 3;
  const int mBase = blockIdx.y * GBM;
  const int nBase = blockIdx.x * GBN;
  const int ntiles = K / GBK;

  v8f acc00 = {}, acc01 = {}, acc10 = {}, acc11 = {};

  // prologue: TDM tile 0 into buffer 0 (pad: every 32 dwords add 4 dwords)
  if (w == 0) {
    tdm_load_2d_f16(lds_addr_of(&As[0][0][0]), A + (size_t)mBase * K,
                    GBK, GBM, (unsigned)K, (unsigned)(M - mBase),
                    (unsigned long long)K, 1, 4, 3);
    tdm_load_2d_f16(lds_addr_of(&Bs[0][0][0]), W + (size_t)nBase * K,
                    GBK, GBN, (unsigned)K, (unsigned)(N - nBase),
                    (unsigned long long)K, 1, 4, 3);
    __builtin_amdgcn_s_wait_tensorcnt(0);
  }
  __syncthreads();

  int cur = 0;
  for (int kt = 0; kt < ntiles; ++kt) {
    const int nxt = cur ^ 1;
    if (w == 0 && kt + 1 < ntiles) {   // async prefetch of next K tile
      const size_t koff = (size_t)(kt + 1) * GBK;
      tdm_load_2d_f16(lds_addr_of(&As[nxt][0][0]), A + (size_t)mBase * K + koff,
                      GBK, GBM, (unsigned)(K - koff), (unsigned)(M - mBase),
                      (unsigned long long)K, 1, 4, 3);
      tdm_load_2d_f16(lds_addr_of(&Bs[nxt][0][0]), W + (size_t)nBase * K + koff,
                      GBK, GBN, (unsigned)(K - koff), (unsigned)(N - nBase),
                      (unsigned long long)K, 1, 4, 3);
    }
#pragma unroll
    for (int ks = 0; ks < 2; ++ks) {
      v16h a0 = load_frag_a(&As[cur][wm * 32][0],      GLD, ks * 32);
      v16h a1 = load_frag_a(&As[cur][wm * 32 + 16][0], GLD, ks * 32);
      v16h b0 = load_frag_b(&Bs[cur][wn * 32][0],      GLD, ks * 32);
      v16h b1 = load_frag_b(&Bs[cur][wn * 32 + 16][0], GLD, ks * 32);
      acc00 = wmma_f16(a0, b0, acc00);
      acc01 = wmma_f16(a0, b1, acc01);
      acc10 = wmma_f16(a1, b0, acc10);
      acc11 = wmma_f16(a1, b1, acc11);
    }
    if (w == 0) __builtin_amdgcn_s_wait_tensorcnt(0);
    __syncthreads();
    cur = nxt;
  }

  v8f* accs[2][2] = { { &acc00, &acc01 }, { &acc10, &acc11 } };
  for (int fi = 0; fi < 2; ++fi)
    for (int fj = 0; fj < 2; ++fj)
      for (int r = 0; r < 8; ++r) {
        int row = mBase + wm * 32 + fi * 16 + r + 8 * hi;
        int col = nBase + wn * 32 + fj * 16 + ln;
        if (row < M) {
          float v = (*accs[fi][fj])[r];
          if (bias) v += bias[col];
          if (epi == EPI_GELU) {
            v = gelu_tanh(v);
          } else if (epi == EPI_RES) {
            float g = gate ? gate[(row / rows_per_batch) * gate_ld + col] : 1.0f;
            v = resid[(size_t)row * N + col] + g * v;
          }
          if (C)   C[(size_t)row * N + col]   = v;
          if (C16) C16[(size_t)row * N + col] = (_Float16)v;
        }
      }
}

// ---------------- Flash attention (WMMA QK^T and PV, TDM-staged Q/K) --------
#define AHDP 96          // head dim padded for 3 K-steps (TDM zero-fills 72..95)
#define AVLD 72          // key stride in Vt / Pb (144B, 16B multiple)

__global__ __launch_bounds__(128) void attn_kernel(
    const _Float16* __restrict__ Q, int qld,
    const _Float16* __restrict__ Kp, const _Float16* __restrict__ Vp, int kvld,
    int qlen, int kvlen,
    _Float16* __restrict__ O, int old_) {
  __shared__ __attribute__((aligned(16))) _Float16 Qs[64][AHDP];
  __shared__ __attribute__((aligned(16))) _Float16 Ks[64][AHDP];
  __shared__ __attribute__((aligned(16))) _Float16 Vt[80][AVLD];  // [dim][key]
  __shared__ __attribute__((aligned(16))) _Float16 Pb[64][AVLD];
  __shared__ float Sb[64][64];
  __shared__ float rowM[64], rowL[64], rowC[64];

  const int tid  = threadIdx.x;
  const int w    = tid >> 5;
  const int lane = tid & 31;
  const int hi   = lane >> 4;
  const int ln   = lane & 15;
  const int b  = blockIdx.z;
  const int h  = blockIdx.y;
  const int qb = blockIdx.x * 64;

  const _Float16* qbase = Q  + (size_t)b * qlen  * qld  + h * DIT_HD;
  const _Float16* kbase = Kp + (size_t)b * kvlen * kvld + h * DIT_HD;
  const _Float16* vbase = Vp + (size_t)b * kvlen * kvld + h * DIT_HD;

  // Q tile via TDM: tensor_dim0=72 zero-fills cols 72..95; no LDS pad
  if (w == 0) {
    tdm_load_2d_f16(lds_addr_of(&Qs[0][0]), qbase + (size_t)qb * qld,
                    AHDP, 64, DIT_HD, (unsigned)(qlen - qb),
                    (unsigned long long)qld, 0, 0, 0);
  }
  if (tid < 64) { rowM[tid] = -1e30f; rowL[tid] = 0.f; }
  v8f oacc[5] = {};
  if (w == 0) __builtin_amdgcn_s_wait_tensorcnt(0);
  __syncthreads();

  const float scale = 0.11785113019775793f;  // 72^-0.5

  for (int kt = 0; kt < kvlen; kt += 64) {
    // async K tile (rows >= kvlen-kt zero-filled by TDM)
    if (w == 0) {
      tdm_load_2d_f16(lds_addr_of(&Ks[0][0]), kbase + (size_t)kt * kvld,
                      AHDP, 64, DIT_HD, (unsigned)(kvlen - kt),
                      (unsigned long long)kvld, 0, 0, 0);
    }
    // overlap: stage V transposed (TDM cannot transpose)
    for (int idx = tid; idx < 80 * 64; idx += 128) {
      int d = idx >> 6, kk = idx & 63;
      _Float16 v = (_Float16)0.f;
      if (d < DIT_HD && (kt + kk) < kvlen) v = vbase[(size_t)(kt + kk) * kvld + d];
      Vt[d][kk] = v;
    }
    if (w == 0) __builtin_amdgcn_s_wait_tensorcnt(0);
    __syncthreads();

    // S = scale * Q K^T (wave w owns query rows w*16..w*16+15)
    for (int fj = 0; fj < 4; ++fj) {
      v8f s = {};
#pragma unroll
      for (int ks = 0; ks < 3; ++ks) {
        v16h a  = load_frag_a(&Qs[w * 16][0],  AHDP, ks * 32);
        v16h bb = load_frag_b(&Ks[fj * 16][0], AHDP, ks * 32);
        s = wmma_f16(a, bb, s);
      }
      for (int r = 0; r < 8; ++r)
        Sb[w * 16 + r + 8 * hi][fj * 16 + ln] = s[r] * scale;
    }
    __syncthreads();

    // online softmax: 2 threads per row, 32 cols each
    {
      int rid = tid >> 1;
      int c0  = (tid & 1) * 32;
      float mloc = -1e30f;
      for (int cc = c0; cc < c0 + 32; ++cc)
        if (kt + cc < kvlen) mloc = fmaxf(mloc, Sb[rid][cc]);
      mloc = fmaxf(mloc, __shfl_xor(mloc, 1, 32));
      float mprev = rowM[rid];
      float mnew  = fmaxf(mprev, mloc);
      float corr  = __expf(mprev - mnew);
      float lsum  = 0.f;
      for (int cc = c0; cc < c0 + 32; ++cc) {
        float p = 0.f;
        if (kt + cc < kvlen) p = __expf(Sb[rid][cc] - mnew);
        Pb[rid][cc] = (_Float16)p;
        lsum += p;
      }
      lsum += __shfl_xor(lsum, 1, 32);
      if ((tid & 1) == 0) {
        rowM[rid] = mnew;
        rowL[rid] = rowL[rid] * corr + lsum;
        rowC[rid] = corr;
      }
    }
    __syncthreads();

    // rescale O, then O += P V
    for (int fd = 0; fd < 5; ++fd)
      for (int r = 0; r < 8; ++r)
        oacc[fd][r] *= rowC[w * 16 + r + 8 * hi];
#pragma unroll
    for (int ks = 0; ks < 2; ++ks) {
      v16h a = load_frag_a(&Pb[w * 16][0], AVLD, ks * 32);
      for (int fd = 0; fd < 5; ++fd) {
        v16h bb = load_frag_b(&Vt[fd * 16][0], AVLD, ks * 32);
        oacc[fd] = wmma_f16(a, bb, oacc[fd]);
      }
    }
    __syncthreads();
  }

  for (int fd = 0; fd < 5; ++fd)
    for (int r = 0; r < 8; ++r) {
      int row = w * 16 + r + 8 * hi;
      int d   = fd * 16 + ln;
      if (d < DIT_HD) {
        float v = oacc[fd][r] / rowL[row];
        O[((size_t)b * qlen + qb + row) * old_ + h * DIT_HD + d] = (_Float16)v;
      }
    }
}

// ---------------------------------------------------------------------------
extern "C" void kernel_launch(void* const* d_in, const int* in_sizes, int n_in,
                              void* d_out, int out_size, void* d_ws, size_t ws_size,
                              hipStream_t stream) {
  (void)in_sizes; (void)n_in; (void)out_size; (void)ws_size;
  const float* x        = (const float*)d_in[0];
  const float* c        = (const float*)d_in[1];
  const float* t_emb    = (const float*)d_in[2];
  const float* qkv_w    = (const float*)d_in[3];
  const float* qkv_b    = (const float*)d_in[4];
  const float* proj_w   = (const float*)d_in[5];
  const float* proj_b   = (const float*)d_in[6];
  const float* to_q_w   = (const float*)d_in[7];
  const float* to_k_w   = (const float*)d_in[8];
  const float* to_v_w   = (const float*)d_in[9];
  const float* to_out_w = (const float*)d_in[10];
  const float* to_out_b = (const float*)d_in[11];
  const float* mlp_w1   = (const float*)d_in[12];
  const float* mlp_b1   = (const float*)d_in[13];
  const float* mlp_w2   = (const float*)d_in[14];
  const float* mlp_b2   = (const float*)d_in[15];
  const float* ada_w    = (const float*)d_in[16];
  const float* ada_b    = (const float*)d_in[17];
  float* out = (float*)d_out;

  // workspace carve (256B aligned chunks)
  char* wp = (char*)d_ws;
  auto allocB = [&](size_t bytes) {
    void* p = (void*)wp;
    wp += (bytes + 255) & ~(size_t)255;
    return p;
  };
  auto allocF = [&](size_t n) { return (float*)allocB(n * 4); };
  auto allocH = [&](size_t n) { return (_Float16*)allocB(n * 2); };

  float*    mod    = allocF((size_t)DIT_B * DIT_MODW);
  float*    x1     = allocF((size_t)DIT_MTOK * DIT_H);
  float*    x2     = allocF((size_t)DIT_MTOK * DIT_H);
  _Float16* w_qkv  = allocH((size_t)3 * DIT_H * DIT_H);
  _Float16* w_proj = allocH((size_t)DIT_H * DIT_H);
  _Float16* w_toq  = allocH((size_t)DIT_H * DIT_H);
  _Float16* w_tok  = allocH((size_t)DIT_H * DIT_CTX);
  _Float16* w_tov  = allocH((size_t)DIT_H * DIT_CTX);
  _Float16* w_toou = allocH((size_t)DIT_H * DIT_H);
  _Float16* w_m1   = allocH((size_t)DIT_MLP * DIT_H);
  _Float16* w_m2   = allocH((size_t)DIT_H * DIT_MLP);
  _Float16* c16    = allocH((size_t)DIT_CROWS * DIT_CTX);
  _Float16* h16    = allocH((size_t)DIT_MTOK * DIT_H);
  _Float16* qkv16  = allocH((size_t)DIT_MTOK * 3 * DIT_H);
  _Float16* ao16   = allocH((size_t)DIT_MTOK * DIT_H);
  _Float16* x116   = allocH((size_t)DIT_MTOK * DIT_H);
  _Float16* q216   = allocH((size_t)DIT_MTOK * DIT_H);
  _Float16* k216   = allocH((size_t)DIT_CROWS * DIT_H);
  _Float16* v216   = allocH((size_t)DIT_CROWS * DIT_H);
  _Float16* mlp16  = allocH((size_t)DIT_MTOK * DIT_MLP);

  auto cvt = [&](const float* s, _Float16* d, size_t n) {
    int n4 = (int)(n / 4);
    cvt_f16_kernel<<<(n4 + 255) / 256, 256, 0, stream>>>(s, d, n4);
  };
  // 0) one-time f16 conversions (weights + context)
  cvt(qkv_w,    w_qkv,  (size_t)3 * DIT_H * DIT_H);
  cvt(proj_w,   w_proj, (size_t)DIT_H * DIT_H);
  cvt(to_q_w,   w_toq,  (size_t)DIT_H * DIT_H);
  cvt(to_k_w,   w_tok,  (size_t)DIT_H * DIT_CTX);
  cvt(to_v_w,   w_tov,  (size_t)DIT_H * DIT_CTX);
  cvt(to_out_w, w_toou, (size_t)DIT_H * DIT_H);
  cvt(mlp_w1,   w_m1,   (size_t)DIT_MLP * DIT_H);
  cvt(mlp_w2,   w_m2,   (size_t)DIT_H * DIT_MLP);
  cvt(c,        c16,    (size_t)DIT_CROWS * DIT_CTX);

  const dim3 gthr(512);
  // 1) adaLN modulation
  ada_kernel<<<(DIT_B * DIT_MODW) / 256, 256, 0, stream>>>(t_emb, ada_w, ada_b, mod);
  // 2) h1 = modulate(ln(x), sh_msa, sc_msa)  (f16)
  ln_mod_kernel<<<DIT_MTOK, 128, 0, stream>>>(x, mod, 0, DIT_H, h16);
  // 3) qkv = h1 @ qkv_w^T + b  (f16)
  gemm_kernel<<<dim3((3 * DIT_H) / GBN, DIT_MTOK / GBM), gthr, 0, stream>>>(
      h16, w_qkv, qkv_b, nullptr, qkv16, DIT_MTOK, 3 * DIT_H, DIT_H, EPI_NONE,
      nullptr, nullptr, 0, 1);
  // 4) self attention  (f16)
  attn_kernel<<<dim3(DIT_N / 64, DIT_NH, DIT_B), 128, 0, stream>>>(
      qkv16, 3 * DIT_H, qkv16 + DIT_H, qkv16 + 2 * DIT_H, 3 * DIT_H,
      DIT_N, DIT_N, ao16, DIT_H);
  // 5) x1 = x + g_msa * (ao @ proj_w^T + b)  (f32 + f16 copies)
  gemm_kernel<<<dim3(DIT_H / GBN, DIT_MTOK / GBM), gthr, 0, stream>>>(
      ao16, w_proj, proj_b, x1, x116, DIT_MTOK, DIT_H, DIT_H, EPI_RES,
      x, mod + 2 * DIT_H, DIT_MODW, DIT_N);
  // 6) cross-attn projections (f16)
  gemm_kernel<<<dim3(DIT_H / GBN, DIT_MTOK / GBM), gthr, 0, stream>>>(
      x116, w_toq, nullptr, nullptr, q216, DIT_MTOK, DIT_H, DIT_H, EPI_NONE,
      nullptr, nullptr, 0, 1);
  gemm_kernel<<<dim3(DIT_H / GBN, (DIT_CROWS + GBM - 1) / GBM), gthr, 0, stream>>>(
      c16, w_tok, nullptr, nullptr, k216, DIT_CROWS, DIT_H, DIT_CTX, EPI_NONE,
      nullptr, nullptr, 0, 1);
  gemm_kernel<<<dim3(DIT_H / GBN, (DIT_CROWS + GBM - 1) / GBM), gthr, 0, stream>>>(
      c16, w_tov, nullptr, nullptr, v216, DIT_CROWS, DIT_H, DIT_CTX, EPI_NONE,
      nullptr, nullptr, 0, 1);
  // 7) cross attention (kv_len = 77)
  attn_kernel<<<dim3(DIT_N / 64, DIT_NH, DIT_B), 128, 0, stream>>>(
      q216, DIT_H, k216, v216, DIT_H, DIT_N, DIT_CTOK, ao16, DIT_H);
  // 8) x2 = x1 + (ao @ to_out_w^T + b)
  gemm_kernel<<<dim3(DIT_H / GBN, DIT_MTOK / GBM), gthr, 0, stream>>>(
      ao16, w_toou, to_out_b, x2, nullptr, DIT_MTOK, DIT_H, DIT_H, EPI_RES,
      x1, nullptr, 0, DIT_N);
  // 9) h2 = modulate(ln(x2), sh_mlp, sc_mlp)  (f16)
  ln_mod_kernel<<<DIT_MTOK, 128, 0, stream>>>(x2, mod, 3 * DIT_H, 4 * DIT_H, h16);
  // 10) mlp hidden = gelu(h2 @ w1^T + b1)  (f16)
  gemm_kernel<<<dim3(DIT_MLP / GBN, DIT_MTOK / GBM), gthr, 0, stream>>>(
      h16, w_m1, mlp_b1, nullptr, mlp16, DIT_MTOK, DIT_MLP, DIT_H, EPI_GELU,
      nullptr, nullptr, 0, 1);
  // 11) out = x2 + g_mlp * (hidden @ w2^T + b2)
  gemm_kernel<<<dim3(DIT_H / GBN, DIT_MTOK / GBM), gthr, 0, stream>>>(
      mlp16, w_m2, mlp_b2, out, nullptr, DIT_MTOK, DIT_H, DIT_MLP, EPI_RES,
      x2, mod + 5 * DIT_H, DIT_MODW, DIT_N);
}
